// SelfAttention_87952340287507
// MI455X (gfx1250) — matrix operand order repaired and checked by reading, not compiled, not run on hardware
//
#include <hip/hip_runtime.h>
#include <hip/hip_bf16.h>

typedef __attribute__((ext_vector_type(16))) __bf16 v16bf;
typedef __attribute__((ext_vector_type(8)))  __bf16 v8bf;
typedef __attribute__((ext_vector_type(8)))  float  v8f;
typedef __attribute__((ext_vector_type(4)))  unsigned int v4u;
typedef __attribute__((ext_vector_type(8)))  int v8i;
typedef __attribute__((ext_vector_type(4)))  int v4i;

#if defined(__has_builtin)
#  if __has_builtin(__builtin_amdgcn_tensor_load_to_lds)
#    define USE_TDM 1
#  endif
#endif
#ifndef USE_TDM
#  define USE_TDM 0
#endif

namespace {
constexpr int NB = 4;     // batch
constexpr int NC = 512;   // channels
constexpr int NN = 6400;  // H*W
constexpr int ND = 64;    // attention dim (C/8)
constexpr int KP = 72;    // padded LDS row (bf16 elems) for K tiles (32x64)
constexpr int VP = 40;    // padded LDS row (bf16 elems) for V tiles (64x32)
constexpr int NSTEP = NN / 32;  // 200 key steps
}

// ---- helpers -------------------------------------------------------------

// A-matrix (16x32 bf16) per-lane fragment from a row-major bf16 row pointer.
// Lane layout: row M = lane&15; elements 0..15 = K (base+0..15);
// base = (lane<16 ? 0 : 16) is folded into `p` by the caller.
__device__ __forceinline__ v16bf load_a_bf16(const __bf16* p) {
  v8bf lo = *(const v8bf*)(p);
  v8bf hi = *(const v8bf*)(p + 8);
  v16bf a;
#pragma unroll
  for (int k = 0; k < 8; ++k) { a[k] = lo[k]; a[k + 8] = hi[k]; }
  return a;
}

__device__ __forceinline__ v8f wmma_bf16(v16bf a, v16bf b, v8f c) {
  // D = A(16x32) * B(32x16) + C, fp32 accumulate
  return __builtin_amdgcn_wmma_f32_16x16x32_bf16(false, a, false, b, (short)0, c,
                                                 false, false);
}

#if USE_TDM
// Issue one TDM tensor load (2-D tile, groups 2/3 unused -> zero).
__device__ __forceinline__ void tdm_load(v4u g0, v8i g1) {
  v4i z4 = {};
#if __clang_major__ >= 23
  v8i z8 = {};
  __builtin_amdgcn_tensor_load_to_lds(g0, g1, z4, z4, z8, 0);
#else
  __builtin_amdgcn_tensor_load_to_lds(g0, g1, z4, z4, 0);
#endif
}
#endif

// ---- kernel 0: one-shot fp32 -> bf16 weight conversion -------------------
__global__ void __launch_bounds__(256)
cvt_weights_kernel(const float* __restrict__ Wf, const float* __restrict__ Wg,
                   const float* __restrict__ Wh, const float* __restrict__ Wv,
                   __bf16* __restrict__ Wfb, __bf16* __restrict__ Wgb,
                   __bf16* __restrict__ Whb, __bf16* __restrict__ Wvb) {
  const int i = blockIdx.x * 256 + threadIdx.x;
  if (i < ND * NC) {
    Wfb[i] = (__bf16)Wf[i];
    Wgb[i] = (__bf16)Wg[i];
    Whb[i] = (__bf16)Wh[i];
    Wvb[i] = (__bf16)Wv[i];
  }
}

// ---- kernel 1: QKV projections (1x1 convs) -------------------------------
//   Krm = f as [B][N][64] row-major   (A operand of S^T = K*Q^T)
//   Qrm = g as [B][N][64] row-major   (B operand: each query's d contiguous)
//   Vcm = h as [B][64][N] chan-major  (A operand of O^T = V^T*P^T)
__global__ void __launch_bounds__(256)
qkv_proj_kernel(const float* __restrict__ x,
                const __bf16* __restrict__ Wfb, const float* __restrict__ bfv,
                const __bf16* __restrict__ Wgb, const float* __restrict__ bgv,
                const __bf16* __restrict__ Whb, const float* __restrict__ bhv,
                __bf16* __restrict__ Krm, __bf16* __restrict__ Qrm,
                __bf16* __restrict__ Vcm) {
  const int lane = threadIdx.x & 31;
  const int wid  = threadIdx.x >> 5;
  const int b    = blockIdx.y;
  const int n0   = (blockIdx.x * 8 + wid) * 16;
  const int col  = lane & 15;
  const int hg   = lane >> 4;  // half-wave group

  const float* xb = x + (size_t)b * NC * NN;

  v8f zero = {};
  v8f accF[4], accG[4], accH[4];
#pragma unroll
  for (int t = 0; t < 4; ++t) { accF[t] = zero; accG[t] = zero; accH[t] = zero; }

  for (int c0 = 0; c0 < NC; c0 += 32) {
    // B operand: x tile, K rows = channels c0..c0+31, cols = pixels n0..n0+15.
    v16bf bx;
    const float* xcol = xb + (size_t)(c0 + hg * 16) * NN + (n0 + col);
#pragma unroll
    for (int k = 0; k < 16; ++k) bx[k] = (__bf16)xcol[(size_t)k * NN];

#pragma unroll
    for (int t = 0; t < 4; ++t) {
      const int row = t * 16 + col;                 // output channel (M)
      const int ko  = c0 + hg * 16;                 // A K-base for this lane
      accF[t] = wmma_bf16(load_a_bf16(Wfb + row * NC + ko), bx, accF[t]);
      accG[t] = wmma_bf16(load_a_bf16(Wgb + row * NC + ko), bx, accG[t]);
      accH[t] = wmma_bf16(load_a_bf16(Whb + row * NC + ko), bx, accH[t]);
    }
  }

  const int n = n0 + col;
  __bf16* krow = Krm + ((size_t)b * NN + n) * ND;
  __bf16* qrow = Qrm + ((size_t)b * NN + n) * ND;
#pragma unroll
  for (int t = 0; t < 4; ++t) {
    const int ob = t * 16 + hg * 8;
#pragma unroll
    for (int r = 0; r < 8; ++r) {
      krow[ob + r] = (__bf16)(accF[t][r] + bfv[ob + r]);
      qrow[ob + r] = (__bf16)(accG[t][r] + bgv[ob + r]);
      Vcm[((size_t)b * ND + ob + r) * NN + n] = (__bf16)(accH[t][r] + bhv[ob + r]);
    }
  }
}

// ---- kernel 2: flash attention, K/V tiles staged via TDM into LDS --------
// Block = 8 waves x 16 queries.  Per 32-key step, K (32x64) and V (64x32)
// bf16 tiles land in double-buffered LDS.  TDM path: wave 0 issues two
// tensor_load_to_lds descriptors for tile s+1 (async during compute of tile
// s), waits TENSORcnt, then the block barrier publishes the buffer.  TDM
// padding (4 DWORDs per row) reproduces the bank-conflict-free row pitches.
__global__ void __launch_bounds__(256)
flash_attn_kernel(const __bf16* __restrict__ Krm, const __bf16* __restrict__ Qrm,
                  const __bf16* __restrict__ Vcm, __bf16* __restrict__ Orm) {
  __shared__ __bf16 kbuf[2][32 * KP];
  __shared__ __bf16 vbuf[2][64 * VP];

  const int tid  = threadIdx.x;
  const int lane = tid & 31;
  const int wid  = tid >> 5;
  const int b    = blockIdx.y;
  const int q0   = (blockIdx.x * 8 + wid) * 16;
  const int col  = lane & 15;
  const int hg   = lane >> 4;

  const __bf16* kb = Krm + (size_t)b * NN * ND;
  const __bf16* vb = Vcm + (size_t)b * ND * NN;

#if USE_TDM
  // D# group1 (ISA 08 §8.4): data_size=2B, pad_enable, pad_interval/amount,
  // tensor_dim0/1, tile_dim0/1, tensor_dim0_stride.
  // K tile: rows of 64 elems (128B = 32 DW) + 4 DW pad -> KP=72 elem pitch.
  const v8i kg1 = { (int)((1u << 16) | (1u << 20) | (4u << 22) | (3u << 25)),
                    (int)(64u << 16),          // tensor_dim0 = 64
                    (int)((unsigned)NN << 16), // tensor_dim1 = N
                    (int)(64u << 16),          // tile_dim0 = 64
                    32,                        // tile_dim1 = 32
                    64,                        // tensor_dim0_stride = 64
                    0, 0 };
  // V tile: rows of 32 elems (64B = 16 DW) + 4 DW pad -> VP=40 elem pitch.
  const v8i vg1 = { (int)((1u << 16) | (1u << 20) | (3u << 22) | (3u << 25)),
                    (int)((unsigned)NN << 16), // tensor_dim0 = N
                    (int)(64u << 16),          // tensor_dim1 = 64
                    (int)(32u << 16),          // tile_dim0 = 32
                    64,                        // tile_dim1 = 64
                    NN,                        // tensor_dim0_stride = N
                    0, 0 };
  const unsigned int klds[2] = { (unsigned int)(unsigned long long)&kbuf[0][0],
                                 (unsigned int)(unsigned long long)&kbuf[1][0] };
  const unsigned int vlds[2] = { (unsigned int)(unsigned long long)&vbuf[0][0],
                                 (unsigned int)(unsigned long long)&vbuf[1][0] };
  const unsigned long long kga0 = (unsigned long long)kb;
  const unsigned long long vga0 = (unsigned long long)vb;

  if (wid == 0) {  // prologue: fill buffer 0 with tile 0
    v4u kg0 = { 1u, klds[0], (unsigned int)kga0,
                (unsigned int)(kga0 >> 32) | 0x80000000u };  // type=2
    v4u vg0 = { 1u, vlds[0], (unsigned int)vga0,
                (unsigned int)(vga0 >> 32) | 0x80000000u };
    tdm_load(kg0, kg1);
    tdm_load(vg0, vg1);
    __builtin_amdgcn_s_wait_tensorcnt(0);
  }
  __syncthreads();
#else
  // Fallback: per-thread register staging (2 x 16B each of K and V).
  const int krow = tid >> 3, koff = (tid & 7) * 8;
  const int vrow = tid >> 2, voff = (tid & 3) * 8;
  uint4 kreg = *(const uint4*)(kb + (size_t)krow * ND + koff);
  uint4 vreg = *(const uint4*)(vb + (size_t)vrow * NN + voff);
#endif

  // Loop-invariant query operand (B matrix of S^T): col = query, K-dim = d.
  const __bf16* qrow = Qrm + ((size_t)b * NN + q0 + col) * ND + hg * 16;
  const v16bf qb0 = *(const v16bf*)(qrow);        // d 0..31
  const v16bf qb1 = *(const v16bf*)(qrow + 32);   // d 32..63

  v8f zero = {};
  v8f oacc[4];
#pragma unroll
  for (int t = 0; t < 4; ++t) oacc[t] = zero;
  float m_run = -3.0e38f;
  float l_run = 0.0f;

  for (int s = 0; s < NSTEP; ++s) {
    const int buf = s & 1;
#if USE_TDM
    if (wid == 0 && s + 1 < NSTEP) {  // async-fill the other buffer
      const int nb = (s + 1) & 1;
      const unsigned long long kga = kga0 + (unsigned long long)(s + 1) * 32 * ND * 2;
      const unsigned long long vga = vga0 + (unsigned long long)(s + 1) * 32 * 2;
      v4u kg0 = { 1u, klds[nb], (unsigned int)kga,
                  (unsigned int)(kga >> 32) | 0x80000000u };
      v4u vg0 = { 1u, vlds[nb], (unsigned int)vga,
                  (unsigned int)(vga >> 32) | 0x80000000u };
      tdm_load(kg0, kg1);
      tdm_load(vg0, vg1);
    }
#else
    *(uint4*)(&kbuf[buf][krow * KP + koff]) = kreg;
    *(uint4*)(&vbuf[buf][vrow * VP + voff]) = vreg;
    __syncthreads();
    if (s + 1 < NSTEP) {
      const int m1 = (s + 1) * 32;
      kreg = *(const uint4*)(kb + (size_t)(m1 + krow) * ND + koff);
      vreg = *(const uint4*)(vb + (size_t)vrow * NN + m1 + voff);
    }
#endif

    // scores^T for two 16-key tiles; K-dim = d (2 x k=32)
    v8f sc[2];
#pragma unroll
    for (int kt = 0; kt < 2; ++kt) {
      const __bf16* kr = &kbuf[buf][(kt * 16 + col) * KP + hg * 16];
      v8f acc = zero;
      acc = wmma_bf16(load_a_bf16(kr),      qb0, acc);
      acc = wmma_bf16(load_a_bf16(kr + 32), qb1, acc);
      sc[kt] = acc;
    }

    // Online softmax. Lane holds keys {32s + kt*16 + hg*8 + r} of its query.
    float smax = -3.0e38f;
#pragma unroll
    for (int kt = 0; kt < 2; ++kt)
#pragma unroll
      for (int r = 0; r < 8; ++r) smax = fmaxf(smax, sc[kt][r]);
    smax = fmaxf(smax, __shfl_xor(smax, 16, 32));   // partner lane = same query
    const float m_new = fmaxf(m_run, smax);
    const float corr  = __expf(m_run - m_new);

    float p0[8], p1[8];
    float ssum = 0.0f;
#pragma unroll
    for (int r = 0; r < 8; ++r) {
      p0[r] = __expf(sc[0][r] - m_new);
      p1[r] = __expf(sc[1][r] - m_new);
      ssum += p0[r] + p1[r];
    }
    ssum += __shfl_xor(ssum, 16, 32);
    l_run = l_run * corr + ssum;
    m_run = m_new;

#pragma unroll
    for (int t = 0; t < 4; ++t)
#pragma unroll
      for (int r = 0; r < 8; ++r) oacc[t][r] *= corr;

    // Build P^T B-operand: lanes 0-15 need keys +0..15, lanes 16-31 +16..31.
    v16bf pB;
#pragma unroll
    for (int r = 0; r < 8; ++r) {
      const float o0 = __shfl_xor(p0[r], 16, 32);
      const float o1 = __shfl_xor(p1[r], 16, 32);
      pB[r]     = (__bf16)(hg ? o1 : p0[r]);
      pB[r + 8] = (__bf16)(hg ? p1[r] : o0);
    }

    // O^T += V^T * P^T  (4 d-tiles of 16)
#pragma unroll
    for (int t = 0; t < 4; ++t) {
      const __bf16* vr = &vbuf[buf][(t * 16 + col) * VP + hg * 16];
      oacc[t] = wmma_bf16(load_a_bf16(vr), pB, oacc[t]);
    }

#if USE_TDM
    if (wid == 0 && s + 1 < NSTEP) __builtin_amdgcn_s_wait_tensorcnt(0);
    __syncthreads();  // publish next buffer / retire reads of current one
#endif
  }

  // Normalize (l_run is per-query == per-lane) and store O row-major bf16.
  const float inv = 1.0f / l_run;
  __bf16* orow = Orm + ((size_t)b * NN + q0 + col) * ND;
#pragma unroll
  for (int t = 0; t < 4; ++t)
#pragma unroll
    for (int r = 0; r < 8; ++r)
      orow[t * 16 + hg * 8 + r] = (__bf16)(oacc[t][r] * inv);
}

// ---- kernel 3: output projection + bias + residual ----------------------
__global__ void __launch_bounds__(256)
out_proj_kernel(const float* __restrict__ x, const __bf16* __restrict__ Wvb,
                const float* __restrict__ bvv, const __bf16* __restrict__ Orm,
                float* __restrict__ y) {
  const int lane = threadIdx.x & 31;
  const int wid  = threadIdx.x >> 5;
  const int b    = blockIdx.y;
  const int n0   = (blockIdx.x * 8 + wid) * 16;
  const int col  = lane & 15;
  const int hg   = lane >> 4;

  const __bf16* orow = Orm + ((size_t)b * NN + n0 + col) * ND + hg * 16;
  const v16bf ob0 = *(const v16bf*)(orow);        // d 0..31
  const v16bf ob1 = *(const v16bf*)(orow + 32);   // d 32..63
  v8f zero = {};

  for (int mt = 0; mt < 32; ++mt) {
    const __bf16* wp = Wvb + (mt * 16 + col) * ND + hg * 16;
    v8f acc = zero;
    acc = wmma_bf16(load_a_bf16(wp),      ob0, acc);
    acc = wmma_bf16(load_a_bf16(wp + 32), ob1, acc);

    const int cb = mt * 16 + hg * 8;
#pragma unroll
    for (int r = 0; r < 8; ++r) {
      const size_t idx = ((size_t)b * NC + cb + r) * NN + n0 + col;
      y[idx] = x[idx] + bvv[cb + r] + acc[r];
    }
  }
}

// ---- launch --------------------------------------------------------------
extern "C" void kernel_launch(void* const* d_in, const int* in_sizes, int n_in,
                              void* d_out, int out_size, void* d_ws, size_t ws_size,
                              hipStream_t stream) {
  (void)in_sizes; (void)n_in; (void)out_size; (void)ws_size;
  const float* x   = (const float*)d_in[0];
  const float* Wf  = (const float*)d_in[1];
  const float* bfv = (const float*)d_in[2];
  const float* Wg  = (const float*)d_in[3];
  const float* bgv = (const float*)d_in[4];
  const float* Wh  = (const float*)d_in[5];
  const float* bhv = (const float*)d_in[6];
  const float* Wvm = (const float*)d_in[7];
  const float* bvv = (const float*)d_in[8];
  float* y = (float*)d_out;

  const size_t matElems = (size_t)NB * NN * ND;
  const size_t wElems   = (size_t)ND * NC;
  __bf16* Krm = (__bf16*)d_ws;
  __bf16* Qrm = Krm + matElems;
  __bf16* Vcm = Qrm + matElems;
  __bf16* Orm = Vcm + matElems;
  __bf16* Wfb = Orm + matElems;
  __bf16* Wgb = Wfb + wElems;
  __bf16* Whb = Wgb + wElems;
  __bf16* Wvb = Whb + wElems;

  dim3 blk(256);                 // 8 wave32 per block
  dim3 grd(NN / (16 * 8), NB);   // 50 x 4: one wave per 16-pixel tile

  cvt_weights_kernel<<<dim3((ND * NC + 255) / 256), blk, 0, stream>>>(
      Wf, Wg, Wh, Wvm, Wfb, Wgb, Whb, Wvb);
  qkv_proj_kernel<<<grd, blk, 0, stream>>>(x, Wfb, bfv, Wgb, bgv, Whb, bhv,
                                           Krm, Qrm, Vcm);
  flash_attn_kernel<<<grd, blk, 0, stream>>>(Krm, Qrm, Vcm, Orm);
  out_proj_kernel<<<grd, blk, 0, stream>>>(x, Wvb, bvv, Orm, y);
}
